// BondUpdate_53549652246917
// MI455X (gfx1250) — matrix operand order repaired and checked by reading, not compiled, not run on hardware
//
#include <hip/hip_runtime.h>
#include <hip/hip_bf16.h>
#include <stdint.h>

// ---------- CDNA5 WMMA types ----------
typedef __attribute__((ext_vector_type(16))) __bf16 bf16x16;
typedef __attribute__((ext_vector_type(8)))  float  f32x8;

union FragU { bf16x16 v; uint4 q[2]; };

__device__ __forceinline__ unsigned short f2bf(float f) {
  unsigned int u = __builtin_bit_cast(unsigned int, f);
  u += 0x7fffu + ((u >> 16) & 1u);          // round-to-nearest-even
  return (unsigned short)(u >> 16);
}

// A fragment 16x32 bf16 from LDS (ISA 7.12.2 16-bit A layout):
// lane<16 -> row=lane, K-chunks {k0..k0+7} and {k0+16..k0+23}
// lane>=16 -> row=lane-16, K-chunks {k0+8..k0+15} and {k0+24..k0+31}
__device__ __forceinline__ bf16x16 load_a_frag(const unsigned short* base, int stride,
                                               int row0, int k0, int lane) {
  const unsigned short* p = base + (size_t)(row0 + (lane & 15)) * stride
                                 + k0 + ((lane >> 4) << 3);
  FragU f;
  f.q[0] = *reinterpret_cast<const uint4*>(p);
  f.q[1] = *reinterpret_cast<const uint4*>(p + 16);
  return f.v;
}

// B fragment 32x16 bf16 from pre-packed global weights: 1KB tile, lane-major.
__device__ __forceinline__ bf16x16 load_b_frag(const unsigned short* wp, int KT,
                                               int nt, int kt, int lane) {
  const unsigned short* p = wp + (((size_t)nt * KT + kt) << 9) + (size_t)lane * 16;
  FragU f;
  f.q[0] = *reinterpret_cast<const uint4*>(p);
  f.q[1] = *reinterpret_cast<const uint4*>(p + 8);
  return f.v;
}

// ---------- weight repack: f32 [K x N] row-major -> bf16 B-fragment tiles ----------
// B layout (matches A): lanes 0-15 hold K=0..7,16..23; lanes 16-31 hold K=8..15,24..31;
// lane%16 = column n within the 16-wide tile.
__global__ void pack_weights(const float* __restrict__ W, unsigned short* __restrict__ Wp,
                             int K, int N) {
  int i = blockIdx.x * 256 + threadIdx.x;
  if (i >= K * N) return;
  int k = i / N, n = i - k * N;
  int kt = k >> 5, k_in = k & 31;
  int nt = n >> 4, n_in = n & 15;
  int lane = ((k_in & 8) ? 16 : 0) + n_in;
  int kk = (k_in & 7) + ((k_in & 16) ? 8 : 0);
  size_t dst = (((size_t)nt * (size_t)(K >> 5) + kt) << 9) + (size_t)lane * 16 + kk;
  Wp[dst] = f2bf(W[i]);
}

// ---------- fused MLP layers ----------
// Workgroup = 256 threads = 8 waves; row tile M = 64 (4 row-blocks of 16).
// Each wave owns NT_PER_WAVE n-tiles across ALL 4 row-blocks -> each B frag
// loaded once per workgroup and reused by 4 WMMAs.
template<int KT, int IN_STRIDE, int NT_PER_WAVE, int OUT_STRIDE>
__device__ __forceinline__ void mlp_layer_hidden(const unsigned short* __restrict__ xin,
                                                 const unsigned short* __restrict__ wp,
                                                 const float* __restrict__ bias,
                                                 unsigned short* __restrict__ hout, int tid) {
  const int lane = tid & 31;
  const int wave = tid >> 5;
  const int mh   = (lane >> 4) << 3;        // C-frag: lanes 16-31 -> rows +8
  const int nin  = lane & 15;
  for (int g = 0; g < NT_PER_WAVE; g += 2) {
    const int nt0 = wave * NT_PER_WAVE + g;
    f32x8 acc[2][4];
#pragma unroll
    for (int j = 0; j < 2; ++j)
#pragma unroll
      for (int rb = 0; rb < 4; ++rb)
#pragma unroll
        for (int v = 0; v < 8; ++v) acc[j][rb][v] = 0.0f;

    for (int kt = 0; kt < KT; ++kt) {
      bf16x16 a[4];
#pragma unroll
      for (int rb = 0; rb < 4; ++rb)
        a[rb] = load_a_frag(xin, IN_STRIDE, rb * 16, kt * 32, lane);
#pragma unroll
      for (int j = 0; j < 2; ++j) {
        bf16x16 b = load_b_frag(wp, KT, nt0 + j, kt, lane);
#pragma unroll
        for (int rb = 0; rb < 4; ++rb)
          acc[j][rb] = __builtin_amdgcn_wmma_f32_16x16x32_bf16(
              false, a[rb], false, b, (short)0, acc[j][rb], false, false);
      }
    }
    // bias + ReLU + bf16 store to LDS (C layout: lane<16 M=v, lane>=16 M=v+8)
#pragma unroll
    for (int j = 0; j < 2; ++j) {
      const int n = (nt0 + j) * 16 + nin;
      const float bv = bias[n];
#pragma unroll
      for (int rb = 0; rb < 4; ++rb) {
        const int m0 = rb * 16 + mh;
#pragma unroll
        for (int v = 0; v < 8; ++v) {
          float x = acc[j][rb][v] + bv;
          x = x > 0.0f ? x : 0.0f;
          hout[(size_t)(m0 + v) * OUT_STRIDE + n] = f2bf(x);
        }
      }
    }
  }
}

template<int KT, int IN_STRIDE>
__device__ __forceinline__ void mlp_layer_out(const unsigned short* __restrict__ xin,
                                              const unsigned short* __restrict__ wp,
                                              const float* __restrict__ bias,
                                              float* __restrict__ out,
                                              long bond0, long nB, int tid) {
  const int lane = tid & 31;
  const int wave = tid >> 5;
  const int mh   = (lane >> 4) << 3;
  const int nin  = lane & 15;
  const int nt0  = wave * 2;                // N = 256 -> 2 n-tiles per wave
  f32x8 acc[2][4];
#pragma unroll
  for (int j = 0; j < 2; ++j)
#pragma unroll
    for (int rb = 0; rb < 4; ++rb)
#pragma unroll
      for (int v = 0; v < 8; ++v) acc[j][rb][v] = 0.0f;

  for (int kt = 0; kt < KT; ++kt) {
    bf16x16 a[4];
#pragma unroll
    for (int rb = 0; rb < 4; ++rb)
      a[rb] = load_a_frag(xin, IN_STRIDE, rb * 16, kt * 32, lane);
#pragma unroll
    for (int j = 0; j < 2; ++j) {
      bf16x16 b = load_b_frag(wp, KT, nt0 + j, kt, lane);
#pragma unroll
      for (int rb = 0; rb < 4; ++rb)
        acc[j][rb] = __builtin_amdgcn_wmma_f32_16x16x32_bf16(
            false, a[rb], false, b, (short)0, acc[j][rb], false, false);
    }
  }
#pragma unroll
  for (int j = 0; j < 2; ++j) {
    const int n = (nt0 + j) * 16 + nin;
    const float bv = bias[n];
#pragma unroll
    for (int rb = 0; rb < 4; ++rb) {
      const int m0 = rb * 16 + mh;
#pragma unroll
      for (int v = 0; v < 8; ++v) {
        const long e = bond0 + m0 + v;
        if (e < nB) {
          float x = acc[j][rb][v] + bv;
          x = x > 0.0f ? x : 0.0f;
          out[e * 256 + n] = x;
        }
      }
    }
  }
}

// ---------- fused kernel: gather -> L1 -> L2 -> L3 ----------
// LDS: bufA = 64 x 904 bf16 (X, padded) reused as 64 x 1032 (H2);
//      bufB = 64 x 1032 bf16 (H1).  Total 264,192 B dynamic LDS (<= 320KB/WGP).
__global__ __launch_bounds__(256) void bond_mlp_kernel(
    const float* __restrict__ sites,  const float* __restrict__ bondsf,
    const float* __restrict__ states, const int* __restrict__ idx1,
    const int* __restrict__ idx2,     const int* __restrict__ g2b,
    const unsigned short* __restrict__ W1p, const float* __restrict__ b1,
    const unsigned short* __restrict__ W2p, const float* __restrict__ b2,
    const unsigned short* __restrict__ W3p, const float* __restrict__ b3,
    float* __restrict__ out, long nB) {
  extern __shared__ __align__(16) unsigned short smem[];
  unsigned short* bufA = smem;                     // X (stride 904) then H2 (stride 1032)
  unsigned short* bufB = smem + (size_t)64 * 1032; // H1 (stride 1032)
  const int tid   = threadIdx.x;
  const long bond0 = (long)blockIdx.x * 64;

  // Gather + f32->bf16: row = [sites[i1] | sites[i2] | bonds | states[g]], 224 float4 chunks
  for (int idx = tid; idx < 64 * 224; idx += 256) {
    const int r = idx / 224;
    const int c = idx - r * 224;
    const long e = bond0 + r;
    float4 valf = make_float4(0.f, 0.f, 0.f, 0.f);
    if (e < nB) {
      if (c < 64)       valf = reinterpret_cast<const float4*>(sites  + (size_t)idx1[e] * 256)[c];
      else if (c < 128) valf = reinterpret_cast<const float4*>(sites  + (size_t)idx2[e] * 256)[c - 64];
      else if (c < 192) valf = reinterpret_cast<const float4*>(bondsf + (size_t)e       * 256)[c - 128];
      else              valf = reinterpret_cast<const float4*>(states + (size_t)g2b[e]  * 128)[c - 192];
    }
    const unsigned int u0 = (unsigned int)f2bf(valf.x) | ((unsigned int)f2bf(valf.y) << 16);
    const unsigned int u1 = (unsigned int)f2bf(valf.z) | ((unsigned int)f2bf(valf.w) << 16);
    *reinterpret_cast<uint2*>(&bufA[(size_t)r * 904 + (size_t)c * 4]) = make_uint2(u0, u1);
  }
  __syncthreads();

  mlp_layer_hidden<28, 904, 8, 1032>(bufA, W1p, b1, bufB, tid);  // 896 -> 1024
  __syncthreads();
  mlp_layer_hidden<32, 1032, 8, 1032>(bufB, W2p, b2, bufA, tid); // 1024 -> 1024
  __syncthreads();
  mlp_layer_out<32, 1032>(bufA, W3p, b3, out, bond0, nB, tid);   // 1024 -> 256
}

// ---------- host ----------
extern "C" void kernel_launch(void* const* d_in, const int* in_sizes, int n_in,
                              void* d_out, int out_size, void* d_ws, size_t ws_size,
                              hipStream_t stream) {
  const float* sites  = (const float*)d_in[0];
  const float* bondsf = (const float*)d_in[1];
  const float* states = (const float*)d_in[2];
  const int*   idx1   = (const int*)d_in[3];
  const int*   idx2   = (const int*)d_in[4];
  const int*   g2b    = (const int*)d_in[5];
  const float* W1     = (const float*)d_in[6];
  const float* b1     = (const float*)d_in[7];
  const float* W2     = (const float*)d_in[8];
  const float* b2     = (const float*)d_in[9];
  const float* W3     = (const float*)d_in[10];
  const float* b3     = (const float*)d_in[11];
  float* out = (float*)d_out;
  const long nB = (long)in_sizes[3];            // N_BONDS

  unsigned short* ws  = (unsigned short*)d_ws;
  unsigned short* W1p = ws;                                  // 896*1024
  unsigned short* W2p = W1p + (size_t)896 * 1024;            // 1024*1024
  unsigned short* W3p = W2p + (size_t)1024 * 1024;           // 1024*256

  pack_weights<<<(896 * 1024 + 255) / 256, 256, 0, stream>>>(W1, W1p, 896, 1024);
  pack_weights<<<(1024 * 1024 + 255) / 256, 256, 0, stream>>>(W2, W2p, 1024, 1024);
  pack_weights<<<(1024 * 256 + 255) / 256, 256, 0, stream>>>(W3, W3p, 1024, 256);

  const int nTiles = (int)((nB + 63) / 64);                  // 3125
  const size_t shmem = 2u * (size_t)64 * 1032 * sizeof(unsigned short); // 264192 B
  bond_mlp_kernel<<<nTiles, 256, shmem, stream>>>(
      sites, bondsf, states, idx1, idx2, g2b,
      W1p, b1, W2p, b2, W3p, b3, out, nB);
}
